// Q4Transformer100M_76879914598598
// MI455X (gfx1250) — compile-verified, matrix operand order checked by reading, not run on hardware
//
#include <hip/hip_runtime.h>
#include <stdint.h>
#include <stddef.h>

// ---------------------------------------------------------------------------
// Q4 transformer (fwd + hand-rolled bwd + SGD) for MI455X / gfx1250.
// GEMMs: v_wmma_f32_16x16x32_bf16, wave32, 8 waves/block, 128x128x32 tile.
// LDS is double-buffered; contiguous staging uses the CDNA5 async
// global->LDS engine (ASYNCcnt) when the builtin is available, overlapping
// HBM/L2 latency with WMMA compute. Weight gradients fuse the SGD update.
// ---------------------------------------------------------------------------

typedef float  v8f   __attribute__((ext_vector_type(8)));
typedef __bf16 v16bf __attribute__((ext_vector_type(16)));
typedef int    v4i   __attribute__((ext_vector_type(4)));

#define NTOK   2048
#define DMODEL 1024
#define FFNDIM 4096
#define NLAYER 8

#if defined(__has_builtin)
# if __has_builtin(__builtin_amdgcn_global_load_async_to_lds_b128)
#  define USE_ASYNC_COPY 1
# endif
#endif
#ifndef USE_ASYNC_COPY
# define USE_ASYNC_COPY 0
#endif

__device__ __forceinline__ unsigned short f2bf(float f) {
  union { float f; unsigned u; } x; x.f = f;
  unsigned r = x.u + 0x7FFFu + ((x.u >> 16) & 1u);   // round-to-nearest-even
  return (unsigned short)(r >> 16);
}
__device__ __forceinline__ float bf2f(unsigned short b) {
  union { unsigned u; float f; } x; x.u = ((unsigned)b) << 16; return x.f;
}

#if USE_ASYNC_COPY
// addrspace(3) pointers are plain LDS byte offsets; generic pointers to
// __shared__ carry the offset in their low 32 bits. The async builtin takes
// typed int4 pointers in addrspace(1)/(3).
__device__ __forceinline__ __attribute__((address_space(3))) v4i* to_lds(void* p) {
  return (__attribute__((address_space(3))) v4i*)(unsigned)(size_t)p;
}
__device__ __forceinline__ __attribute__((address_space(1))) v4i* to_glb(const void* p) {
  return (__attribute__((address_space(1))) v4i*)(size_t)p;
}
__device__ __forceinline__ void wait_async0() {
# if __has_builtin(__builtin_amdgcn_s_wait_asynccnt)
  __builtin_amdgcn_s_wait_asynccnt(0);
# else
  asm volatile("s_wait_asynccnt 0x0" ::: "memory");
# endif
}
#endif

// ---------------------------------------------------------------------------
// Generic bf16 GEMM: C[M,N] = op(A) * op(B), f32 accum via WMMA.
//   transA=0 : Atile[m][k] = A[(bm+m)*lda + k0+k]        (row-major A)
//   transA=1 : Atile[m][k] = A[(k0+k)*lda + bm+m]        (A is src^T)
//   transB=1 : Btile[n][k] = B[(bn+n)*ldb + k0+k]        (X @ W^T fast path)
//   transB=0 : Btile[n][k] = B[(k0+k)*ldb + bn+n]        (X @ W)
// Epilogues:
//   0: outB = bf16(acc)
//   1: outF = acc * alpha
//   2: outB = bf16(relu(acc))
//   3: outB = bf16(aux>0 ? acc : 0)        (relu backward mask)
//   4: outF = wsrc - lr[0]*acc             (fused SGD weight update)
//   5: outB = bf16(acc); outF = acc        (prediction)
// ---------------------------------------------------------------------------
template <int EPI>
__global__ __launch_bounds__(256) void gemm_k(
    const unsigned short* __restrict__ A, int lda, int transA,
    const unsigned short* __restrict__ B, int ldb, int transB,
    int M, int N, int K,
    float* __restrict__ outF, unsigned short* __restrict__ outB,
    const unsigned short* __restrict__ aux,
    const float* __restrict__ wsrc, const float* __restrict__ lr,
    float alpha)
{
  constexpr int BM = 128, BN = 128, BK = 32, LS = 40;  // LS: skewed LDS stride
  __shared__ __align__(16) unsigned short a_s[2][BM * LS];
  __shared__ __align__(16) unsigned short b_s[2][BN * LS];

  const int tid   = threadIdx.x;
  const int lane  = tid & 31;
  const int wave  = tid >> 5;
  const int wm    = wave & 1;        // 2 wave rows  (64 rows each)
  const int wn    = wave >> 1;       // 4 wave cols  (32 cols each)
  const int mloc  = lane & 15;
  const int khalf = lane >> 4;
  const int bm    = blockIdx.y * BM;
  const int bn    = blockIdx.x * BN;

  v8f acc[4][2];
#pragma unroll
  for (int i = 0; i < 4; i++)
#pragma unroll
    for (int j = 0; j < 2; j++)
#pragma unroll
      for (int r = 0; r < 8; r++) acc[i][j][r] = 0.0f;

  float lrv = 0.0f;
  if constexpr (EPI == 4) lrv = lr[0];

  // ---- tile staging (async global->LDS when contiguous) ----
  auto stageA = [&](int buf, int k0) {
    if (!transA) {
#pragma unroll
      for (int c = tid; c < (BM * BK) / 8; c += 256) {   // 512 chunks of 8 halves
        const int m = c >> 2, kc = (c & 3) * 8;
#if USE_ASYNC_COPY
        __builtin_amdgcn_global_load_async_to_lds_b128(
            to_glb(A + (size_t)(bm + m) * lda + k0 + kc),
            to_lds(&a_s[buf][m * LS + kc]), 0, 0);
#else
        uint4 v = *(const uint4*)(A + (size_t)(bm + m) * lda + k0 + kc);
        *(uint4*)&a_s[buf][m * LS + kc] = v;
#endif
      }
    } else {
#pragma unroll
      for (int e = tid; e < BM * BK; e += 256) {
        const int m = e & (BM - 1), k = e >> 7;
        a_s[buf][m * LS + k] = A[(size_t)(k0 + k) * lda + bm + m];
      }
    }
  };
  auto stageB = [&](int buf, int k0) {
    if (transB) {
#pragma unroll
      for (int c = tid; c < (BN * BK) / 8; c += 256) {
        const int n = c >> 2, kc = (c & 3) * 8;
#if USE_ASYNC_COPY
        __builtin_amdgcn_global_load_async_to_lds_b128(
            to_glb(B + (size_t)(bn + n) * ldb + k0 + kc),
            to_lds(&b_s[buf][n * LS + kc]), 0, 0);
#else
        uint4 v = *(const uint4*)(B + (size_t)(bn + n) * ldb + k0 + kc);
        *(uint4*)&b_s[buf][n * LS + kc] = v;
#endif
      }
    } else {
#pragma unroll
      for (int e = tid; e < BN * BK; e += 256) {
        const int n = e & (BN - 1), k = e >> 7;
        b_s[buf][n * LS + k] = B[(size_t)(k0 + k) * ldb + bn + n];
      }
    }
  };

  // prologue: fill buffer 0
  stageA(0, 0);
  stageB(0, 0);

  int curb = 0;
  for (int k0 = 0; k0 < K; k0 += BK) {
#if USE_ASYNC_COPY
    wait_async0();          // my async writes into buf[curb] are done
#endif
    __syncthreads();        // all waves' writes done; prior reads of nxt done

    const int nxt = curb ^ 1;
    if (k0 + BK < K) {      // overlap next-tile staging with this tile's WMMAs
      stageA(nxt, k0 + BK);
      stageB(nxt, k0 + BK);
    }

    // ---- fragments (ISA 7.12.2 16-bit A/B layout) + 8 WMMAs ----
    v16bf af[4], bfr[2];
#pragma unroll
    for (int tm = 0; tm < 4; tm++) {
      const unsigned short* p = &a_s[curb][(wm * 64 + tm * 16 + mloc) * LS];
      ((uint4*)&af[tm])[0] = *(const uint4*)(p + khalf * 8);
      ((uint4*)&af[tm])[1] = *(const uint4*)(p + 16 + khalf * 8);
    }
#pragma unroll
    for (int tn = 0; tn < 2; tn++) {
      const unsigned short* p = &b_s[curb][(wn * 32 + tn * 16 + mloc) * LS];
      ((uint4*)&bfr[tn])[0] = *(const uint4*)(p + khalf * 8);
      ((uint4*)&bfr[tn])[1] = *(const uint4*)(p + 16 + khalf * 8);
    }
#pragma unroll
    for (int tm = 0; tm < 4; tm++)
#pragma unroll
      for (int tn = 0; tn < 2; tn++)
        acc[tm][tn] = __builtin_amdgcn_wmma_f32_16x16x32_bf16(
            false, af[tm], false, bfr[tn], (short)0, acc[tm][tn], false, false);

    curb = nxt;
  }

  // ---- epilogue (C layout: lane -> n = lane&15, vgpr r -> m = r + 8*(lane>>4)) ----
#pragma unroll
  for (int tm = 0; tm < 4; tm++)
#pragma unroll
    for (int tn = 0; tn < 2; tn++)
#pragma unroll
      for (int r = 0; r < 8; r++) {
        const int gr = bm + wm * 64 + tm * 16 + r + 8 * khalf;
        const int gc = bn + wn * 32 + tn * 16 + mloc;
        const size_t o = (size_t)gr * N + gc;
        float v = acc[tm][tn][r];
        if constexpr (EPI == 0)      outB[o] = f2bf(v);
        else if constexpr (EPI == 1) outF[o] = v * alpha;
        else if constexpr (EPI == 2) outB[o] = f2bf(fmaxf(v, 0.0f));
        else if constexpr (EPI == 3) outB[o] = f2bf(bf2f(aux[o]) > 0.0f ? v : 0.0f);
        else if constexpr (EPI == 4) outF[o] = wsrc[o] - lrv * v;
        else { outB[o] = f2bf(v); outF[o] = v; }
      }
}

// ---------------------------------------------------------------------------
// Side kernels
// ---------------------------------------------------------------------------
__global__ __launch_bounds__(256) void absmax_k(const float* __restrict__ w,
                                                size_t n, unsigned* __restrict__ out) {
  __shared__ float red[256];
  float m = 0.0f;
  for (size_t i = (size_t)blockIdx.x * blockDim.x + threadIdx.x; i < n;
       i += (size_t)gridDim.x * blockDim.x)
    m = fmaxf(m, fabsf(w[i]));
  red[threadIdx.x] = m;
  __syncthreads();
  for (int s = 128; s > 0; s >>= 1) {
    if ((int)threadIdx.x < s) red[threadIdx.x] = fmaxf(red[threadIdx.x], red[threadIdx.x + s]);
    __syncthreads();
  }
  if (threadIdx.x == 0) atomicMax(out, __float_as_uint(red[0]));  // bits monotonic for >=0
}

__global__ __launch_bounds__(256) void quant_k(const float* __restrict__ w,
                                               const unsigned* __restrict__ amax,
                                               unsigned short* __restrict__ q, size_t n) {
  const float s = fmaxf(__uint_as_float(*amax) / 7.0f, 1e-5f);
  for (size_t i = (size_t)blockIdx.x * blockDim.x + threadIdx.x; i < n;
       i += (size_t)gridDim.x * blockDim.x)
    q[i] = f2bf(fminf(fmaxf(rintf(w[i] / s), -7.0f), 7.0f) * s);
}

__global__ __launch_bounds__(256) void f32_to_bf16_k(const float* __restrict__ x,
                                                     unsigned short* __restrict__ y, size_t n) {
  for (size_t i = (size_t)blockIdx.x * blockDim.x + threadIdx.x; i < n;
       i += (size_t)gridDim.x * blockDim.x)
    y[i] = f2bf(x[i]);
}

__global__ __launch_bounds__(256) void gradinit_k(const float* __restrict__ pred,
                                                  const float* __restrict__ tgt,
                                                  unsigned short* __restrict__ g, size_t n) {
  for (size_t i = (size_t)blockIdx.x * blockDim.x + threadIdx.x; i < n;
       i += (size_t)gridDim.x * blockDim.x)
    g[i] = f2bf(pred[i] - tgt[i]);
}

// one 256-thread block per 2048-wide row
__global__ __launch_bounds__(256) void softmax_k(const float* __restrict__ S,
                                                 unsigned short* __restrict__ Aout) {
  __shared__ float red[256];
  const int tid = threadIdx.x;
  const float* r = S + (size_t)blockIdx.x * NTOK;
  float v[8];
  float m = -3.4e38f;
#pragma unroll
  for (int j = 0; j < 8; j++) { v[j] = r[tid + 256 * j]; m = fmaxf(m, v[j]); }
  red[tid] = m; __syncthreads();
  for (int s = 128; s > 0; s >>= 1) {
    if (tid < s) red[tid] = fmaxf(red[tid], red[tid + s]);
    __syncthreads();
  }
  const float rm = red[0];
  __syncthreads();
  float sum = 0.0f;
#pragma unroll
  for (int j = 0; j < 8; j++) { v[j] = __expf(v[j] - rm); sum += v[j]; }
  red[tid] = sum; __syncthreads();
  for (int s = 128; s > 0; s >>= 1) {
    if (tid < s) red[tid] += red[tid + s];
    __syncthreads();
  }
  const float inv = 1.0f / red[0];
  unsigned short* o = Aout + (size_t)blockIdx.x * NTOK;
#pragma unroll
  for (int j = 0; j < 8; j++) o[tid + 256 * j] = f2bf(v[j] * inv);
}

// ---------------------------------------------------------------------------
// Host-side dispatch
// ---------------------------------------------------------------------------
static void gemm(hipStream_t st, int epi,
                 const unsigned short* A, int lda, int tA,
                 const unsigned short* B, int ldb, int tB,
                 int M, int N, int K,
                 float* outF, unsigned short* outB,
                 const unsigned short* aux, const float* wsrc,
                 const float* lr, float alpha) {
  dim3 g(N / 128, M / 128), b(256);
  switch (epi) {
    case 0: gemm_k<0><<<g,b,0,st>>>(A,lda,tA,B,ldb,tB,M,N,K,outF,outB,aux,wsrc,lr,alpha); break;
    case 1: gemm_k<1><<<g,b,0,st>>>(A,lda,tA,B,ldb,tB,M,N,K,outF,outB,aux,wsrc,lr,alpha); break;
    case 2: gemm_k<2><<<g,b,0,st>>>(A,lda,tA,B,ldb,tB,M,N,K,outF,outB,aux,wsrc,lr,alpha); break;
    case 3: gemm_k<3><<<g,b,0,st>>>(A,lda,tA,B,ldb,tB,M,N,K,outF,outB,aux,wsrc,lr,alpha); break;
    case 4: gemm_k<4><<<g,b,0,st>>>(A,lda,tA,B,ldb,tB,M,N,K,outF,outB,aux,wsrc,lr,alpha); break;
    default: gemm_k<5><<<g,b,0,st>>>(A,lda,tA,B,ldb,tB,M,N,K,outF,outB,aux,wsrc,lr,alpha); break;
  }
}

extern "C" void kernel_launch(void* const* d_in, const int* in_sizes, int n_in,
                              void* d_out, int out_size, void* d_ws, size_t ws_size,
                              hipStream_t stream) {
  (void)in_sizes; (void)n_in; (void)out_size; (void)ws_size;

  const float* x      = (const float*)d_in[0];
  const float* target = (const float*)d_in[1];
  const float* lr     = (const float*)d_in[2];
  const float* w[3 * NLAYER];
  for (int i = 0; i < 3 * NLAYER; i++) w[i] = (const float*)d_in[3 + i];

  const size_t ND = (size_t)NTOK * DMODEL;   // 2M
  const size_t NF = (size_t)NTOK * FFNDIM;   // 8M
  const size_t NN = (size_t)NTOK * NTOK;     // 4M
  const size_t DD = (size_t)DMODEL * DMODEL; // 1M
  const size_t FD = (size_t)FFNDIM * DMODEL; // 4M

  char* base = (char*)d_ws;
  size_t off = 0;
  auto alloc = [&](size_t bytes) -> void* {
    void* p = base + off;
    off += (bytes + 255) & ~(size_t)255;
    return p;
  };

  unsigned short* hid[NLAYER + 1];
  for (int i = 0; i <= NLAYER; i++) hid[i] = (unsigned short*)alloc(ND * 2);
  unsigned short *ctx[NLAYER], *ffn[NLAYER], *wqq[NLAYER], *w1q[NLAYER], *w2q[NLAYER];
  for (int i = 0; i < NLAYER; i++) ctx[i] = (unsigned short*)alloc(ND * 2);
  for (int i = 0; i < NLAYER; i++) ffn[i] = (unsigned short*)alloc(NF * 2);
  for (int i = 0; i < NLAYER; i++) wqq[i] = (unsigned short*)alloc(DD * 2);
  for (int i = 0; i < NLAYER; i++) w1q[i] = (unsigned short*)alloc(FD * 2);
  for (int i = 0; i < NLAYER; i++) w2q[i] = (unsigned short*)alloc(FD * 2);
  unsigned short* qkv    = (unsigned short*)alloc(ND * 2);
  float*          scores = (float*)alloc(NN * 4);
  unsigned short* attn   = (unsigned short*)alloc(NN * 2);
  unsigned short* gh[2]  = { (unsigned short*)alloc(ND * 2), (unsigned short*)alloc(ND * 2) };
  unsigned short* gffn   = (unsigned short*)alloc(NF * 2);
  unsigned short* gctx   = (unsigned short*)alloc(ND * 2);
  unsigned*       amax   = (unsigned*)alloc(3 * NLAYER * sizeof(unsigned));

  // ---- int4 fake-quant of all 24 weight tensors (bf16 output) ----
  (void)hipMemsetAsync(amax, 0, 3 * NLAYER * sizeof(unsigned), stream);
  for (int t = 0; t < 3 * NLAYER; t++) {
    const size_t n = (t % 3 == 0) ? DD : FD;
    absmax_k<<<256, 256, 0, stream>>>(w[t], n, amax + t);
  }
  for (int t = 0; t < 3 * NLAYER; t++) {
    const int L = t / 3, k = t % 3;
    const size_t n = (k == 0) ? DD : FD;
    unsigned short* dst = (k == 0) ? wqq[L] : (k == 1) ? w1q[L] : w2q[L];
    quant_k<<<256, 256, 0, stream>>>(w[t], amax + t, dst, n);
  }

  f32_to_bf16_k<<<256, 256, 0, stream>>>(x, hid[0], ND);

  const float inv_sqrt_d = 0.03125f;  // 1/sqrt(1024)
  float* pred = (float*)d_out;
  size_t woff[3 * NLAYER];
  {
    size_t o = ND;
    for (int t = 0; t < 3 * NLAYER; t++) { woff[t] = o; o += (t % 3 == 0) ? DD : FD; }
  }

  // ---- forward ----
  for (int i = 0; i < NLAYER; i++) {
    // qkv = hidden @ wq^T
    gemm(stream, 0, hid[i], DMODEL, 0, wqq[i], DMODEL, 1,
         NTOK, DMODEL, DMODEL, nullptr, qkv, nullptr, nullptr, nullptr, 0.f);
    // scores = (qkv @ qkv^T) * inv_sqrt_d   (f32)
    gemm(stream, 1, qkv, DMODEL, 0, qkv, DMODEL, 1,
         NTOK, NTOK, DMODEL, scores, nullptr, nullptr, nullptr, nullptr, inv_sqrt_d);
    softmax_k<<<NTOK, 256, 0, stream>>>(scores, attn);
    // context = attn @ qkv
    gemm(stream, 0, attn, NTOK, 0, qkv, DMODEL, 0,
         NTOK, DMODEL, NTOK, nullptr, ctx[i], nullptr, nullptr, nullptr, 0.f);
    // ffn_h = relu(context @ w1^T)
    gemm(stream, 2, ctx[i], DMODEL, 0, w1q[i], DMODEL, 1,
         NTOK, FFNDIM, DMODEL, nullptr, ffn[i], nullptr, nullptr, nullptr, 0.f);
    // out = ffn_h @ w2^T   (last layer also writes f32 prediction)
    if (i < NLAYER - 1)
      gemm(stream, 0, ffn[i], FFNDIM, 0, w2q[i], FFNDIM, 1,
           NTOK, DMODEL, FFNDIM, nullptr, hid[i + 1], nullptr, nullptr, nullptr, 0.f);
    else
      gemm(stream, 5, ffn[i], FFNDIM, 0, w2q[i], FFNDIM, 1,
           NTOK, DMODEL, FFNDIM, pred, hid[i + 1], nullptr, nullptr, nullptr, 0.f);
  }

  // ---- backward + fused SGD ----
  gradinit_k<<<256, 256, 0, stream>>>(pred, target, gh[0], ND);
  int cur = 0;
  for (int i = NLAYER - 1; i >= 0; i--) {
    unsigned short* g  = gh[cur];
    unsigned short* gn = gh[cur ^ 1];
    // w2 -= lr * (grad_h^T @ ffn_h)
    gemm(stream, 4, g, DMODEL, 1, ffn[i], FFNDIM, 0,
         DMODEL, FFNDIM, NTOK, (float*)d_out + woff[3 * i + 2], nullptr,
         nullptr, w[3 * i + 2], lr, 0.f);
    // g_ffn = relu_mask(ffn_h) * (grad_h @ w2)
    gemm(stream, 3, g, DMODEL, 0, w2q[i], FFNDIM, 0,
         NTOK, FFNDIM, DMODEL, nullptr, gffn, ffn[i], nullptr, nullptr, 0.f);
    // w1 -= lr * (g_ffn^T @ context)
    gemm(stream, 4, gffn, FFNDIM, 1, ctx[i], DMODEL, 0,
         FFNDIM, DMODEL, NTOK, (float*)d_out + woff[3 * i + 1], nullptr,
         nullptr, w[3 * i + 1], lr, 0.f);
    // g_ctx = g_ffn @ w1
    gemm(stream, 0, gffn, FFNDIM, 0, w1q[i], DMODEL, 0,
         NTOK, DMODEL, FFNDIM, nullptr, gctx, nullptr, nullptr, nullptr, 0.f);
    // wq -= lr * (g_ctx^T @ h_in)
    gemm(stream, 4, gctx, DMODEL, 1, hid[i], DMODEL, 0,
         DMODEL, DMODEL, NTOK, (float*)d_out + woff[3 * i], nullptr,
         nullptr, w[3 * i], lr, 0.f);
    // grad_h = g_ctx @ wq
    gemm(stream, 0, gctx, DMODEL, 0, wqq[i], DMODEL, 0,
         NTOK, DMODEL, DMODEL, nullptr, gn, nullptr, nullptr, nullptr, 0.f);
    cur ^= 1;
  }
}
